// DeepSNN_21363167330334
// MI455X (gfx1250) — matrix operand order repaired and checked by reading, not compiled, not run on hardware
//
#include <hip/hip_runtime.h>

typedef __attribute__((ext_vector_type(16))) _Float16 v16h;
typedef __attribute__((ext_vector_type(8)))  _Float16 h8;
typedef __attribute__((ext_vector_type(8)))  float    v8f;
typedef __attribute__((ext_vector_type(4)))  float    f4;

#define SNN_ALPHA 0.9512294245007140f   /* exp(-1/20) */
#define SNN_BETA  0.8187307530779818f   /* exp(-1/5)  */

// ---------------- weight transpose + f32 -> f16 (W[K,N] -> WT[N,K]) -------
__global__ void k_wt(const float* __restrict__ W, _Float16* __restrict__ WT,
                     int K, int N) {
  int idx = blockIdx.x * blockDim.x + threadIdx.x;
  if (idx >= K * N) return;
  int k = idx / N, n = idx - k * N;
  WT[(size_t)n * K + k] = (_Float16)W[idx];
}

// ------------- fragment load helpers --------------------------------------
// A-fragment layout (16-bit, 16x32): lane<16 holds K=0..7,16..23 of row M=lane;
// lane>=16 holds K=8..15,24..31 of row M=lane-16.  Two 16B runs, 16 halves apart.
__device__ __forceinline__ v16h load_a_f32(const float* __restrict__ pa) {
  f4 f0 = *(const f4*)pa;
  f4 f1 = *(const f4*)(pa + 4);
  f4 f2 = *(const f4*)(pa + 16);
  f4 f3 = *(const f4*)(pa + 20);
  v16h a;
  for (int j = 0; j < 4; ++j) {
    a[j]      = (_Float16)f0[j];
    a[4 + j]  = (_Float16)f1[j];
    a[8 + j]  = (_Float16)f2[j];
    a[12 + j] = (_Float16)f3[j];
  }
  return a;
}
// Loads halves p[0..7] and p[off2..off2+7] into one 16-half fragment.
__device__ __forceinline__ v16h load_h16(const _Float16* __restrict__ p, int off2) {
  union { v16h v; h8 h[2]; } u;
  u.h[0] = *(const h8*)p;
  u.h[1] = *(const h8*)(p + off2);
  return u.v;
}

__device__ __forceinline__ v8f wmma_f16(v16h a, v16h b, v8f c) {
  return __builtin_amdgcn_wmma_f32_16x16x32_f16(
      /*neg_a=*/false, a, /*neg_b=*/false, b,
      /*c_mod=*/(short)0, c, /*reuse_a=*/false, /*reuse_b=*/false);
}

// C/D layout: lane<16 -> rows mBase16+0..7 (vgpr i = row i), lane>=16 -> +8..15
__device__ __forceinline__ void store_tile(float* __restrict__ C,
                                           const float* __restrict__ bias,
                                           int N, int mBase16, int nBase16,
                                           int r, int hi, v8f acc) {
  int nC = nBase16 + r;
  float bn = bias[nC];
  int mBase = mBase16 + (hi << 3);
  for (int i = 0; i < 8; ++i)
    C[(size_t)(mBase + i) * N + nC] = acc[i] + bn;
}

// ---------------- WMMA GEMM: C[M,N] = A[M,K] * WT[N,K]^T + bias -----------
// 2x2 register blocking: each wave computes a 32x32 output block with
// 4 accumulators; per K-step: 2 A frags + 2 B frags -> 4 v_wmma.
template<bool A_F32>
__global__ void k_gemm(const void* __restrict__ Av,
                       const _Float16* __restrict__ BT,
                       const float* __restrict__ bias,
                       float* __restrict__ C,
                       int M, int N, int K) {
  const int lane = threadIdx.x & 31;
  const int wave = threadIdx.x >> 5;
  const int nSup = N >> 5;
  int tile = blockIdx.x * 4 + wave;
  if (tile >= (M >> 5) * nSup) return;
  int mt = tile / nSup, nt = tile - mt * nSup;
  int m0 = mt << 5, n0 = nt << 5;

  const int r  = lane & 15;
  const int hi = lane >> 4;
  const int kA = hi << 3;   // 0 or 8
  const int kB = hi << 3;   // 0 or 8

  const float*    Af = (const float*)Av;
  const _Float16* Ah = (const _Float16*)Av;

  v8f acc00 = {}, acc01 = {}, acc10 = {}, acc11 = {};
  for (int k0 = 0; k0 < K; k0 += 32) {
    v16h a0, a1;
    if (A_F32) {
      const float* pa0 = Af + (size_t)(m0 + r) * K + k0 + kA;
      const float* pa1 = Af + (size_t)(m0 + 16 + r) * K + k0 + kA;
      if (k0 + 32 < K) {
        __builtin_prefetch(pa0 + 32, 0, 1);
        __builtin_prefetch(pa1 + 32, 0, 1);
      }
      a0 = load_a_f32(pa0);
      a1 = load_a_f32(pa1);
    } else {
      const _Float16* pa0 = Ah + (size_t)(m0 + r) * K + k0 + kA;
      const _Float16* pa1 = Ah + (size_t)(m0 + 16 + r) * K + k0 + kA;
      a0 = load_h16(pa0, 16);
      a1 = load_h16(pa1, 16);
    }
    v16h b0 = load_h16(BT + (size_t)(n0 + r) * K + k0 + kB, 16);
    v16h b1 = load_h16(BT + (size_t)(n0 + 16 + r) * K + k0 + kB, 16);

    acc00 = wmma_f16(a0, b0, acc00);
    acc01 = wmma_f16(a0, b1, acc01);
    acc10 = wmma_f16(a1, b0, acc10);
    acc11 = wmma_f16(a1, b1, acc11);
  }

  store_tile(C, bias, N, m0,      n0,      r, hi, acc00);
  store_tile(C, bias, N, m0,      n0 + 16, r, hi, acc01);
  store_tile(C, bias, N, m0 + 16, n0,      r, hi, acc10);
  store_tile(C, bias, N, m0 + 16, n0 + 16, r, hi, acc11);
}

// ---------------- LIF scan over T, one thread per (b,h) chain -------------
__global__ void k_lif(const float* __restrict__ xt, float* __restrict__ sp,
                      int T, int H) {
  int idx = blockIdx.x * blockDim.x + threadIdx.x;   // idx = b*H + h
  int b = idx / H, h = idx - b * H;
  const float* px = xt + (size_t)b * T * H + h;
  float* ps = sp + (size_t)b * T * H + h;
  float v = 0.f, cur = 0.f;
  for (int t = 0; t < T; ++t) {
    cur = SNN_BETA * cur + px[(size_t)t * H];
    v = SNN_ALPHA * v + cur;
    float s = (v - 1.0f) >= 0.f ? 1.f : 0.f;
    v = v * (1.f - s);
    ps[(size_t)t * H] = s;
  }
}

// ---------------- LayerNorm + convert to f16, one wave per row ------------
__global__ void k_ln(const float* __restrict__ x, const float* __restrict__ sc,
                     const float* __restrict__ bi, _Float16* __restrict__ y,
                     int H) {
  int row  = (blockIdx.x * blockDim.x + threadIdx.x) >> 5;
  int lane = threadIdx.x & 31;
  int per  = H >> 5;                 // 8 for H=256, 4 for H=128
  const float* px = x + (size_t)row * H;
  float vals[8];
  float s = 0.f;
  for (int j = 0; j < per; ++j) { vals[j] = px[j * 32 + lane]; s += vals[j]; }
  for (int off = 16; off; off >>= 1) s += __shfl_xor(s, off, 32);
  float mean = s / (float)H;
  float var = 0.f;
  for (int j = 0; j < per; ++j) { float d = vals[j] - mean; var += d * d; }
  for (int off = 16; off; off >>= 1) var += __shfl_xor(var, off, 32);
  float rinv = rsqrtf(var / (float)H + 1e-6f);
  _Float16* py = y + (size_t)row * H;
  for (int j = 0; j < per; ++j) {
    int c = j * 32 + lane;
    py[c] = (_Float16)((vals[j] - mean) * rinv * sc[c] + bi[c]);
  }
}

// ---------------- pooled[b] = mean over (t,h); one block per b ------------
__global__ void k_pool(const float* __restrict__ sp, float* __restrict__ out,
                       int T, int H) {
  int b = blockIdx.x;
  const float* p = sp + (size_t)b * T * H;
  int n = T * H;
  float s = 0.f;
  for (int i = threadIdx.x; i < n; i += blockDim.x) s += p[i];
  __shared__ float red[256];
  red[threadIdx.x] = s;
  __syncthreads();
  for (int off = 128; off > 0; off >>= 1) {
    if (threadIdx.x < off) red[threadIdx.x] += red[threadIdx.x + off];
    __syncthreads();
  }
  if (threadIdx.x == 0) out[2 + b] = red[0] / (float)n;
}

// ---------------- head: logits = pooled @ Wc + bc; mean = mean(pooled) ----
__global__ void k_head(const float* __restrict__ Wc, const float* __restrict__ bc,
                       float* __restrict__ out, int B) {
  int t = threadIdx.x;
  __shared__ float l0[64], l1[64], sm[64];
  float p = (t < B) ? out[2 + t] : 0.f;
  l0[t] = p * Wc[t * 2 + 0];
  l1[t] = p * Wc[t * 2 + 1];
  sm[t] = p;
  __syncthreads();
  for (int off = 32; off > 0; off >>= 1) {
    if (t < off) { l0[t] += l0[t + off]; l1[t] += l1[t + off]; sm[t] += sm[t + off]; }
    __syncthreads();
  }
  if (t == 0) {
    out[0]  = l0[0] + bc[0];
    out[1]  = l1[0] + bc[1];
    out[66] = sm[0] / (float)B;
  }
}

extern "C" void kernel_launch(void* const* d_in, const int* in_sizes, int n_in,
                              void* d_out, int out_size, void* d_ws, size_t ws_size,
                              hipStream_t stream) {
  const float* spikes = (const float*)d_in[0];
  const float* W0 = (const float*)d_in[1];
  const float* b0 = (const float*)d_in[2];
  const float* W1 = (const float*)d_in[3];
  const float* b1 = (const float*)d_in[4];
  const float* W2 = (const float*)d_in[5];
  const float* b2 = (const float*)d_in[6];
  const float* ln1s = (const float*)d_in[7];
  const float* ln1b = (const float*)d_in[8];
  const float* ln2s = (const float*)d_in[9];
  const float* ln2b = (const float*)d_in[10];
  const float* Wc = (const float*)d_in[11];
  const float* bc = (const float*)d_in[12];
  float* out = (float*)d_out;

  const int B = 64, T = 512, D = 1024, H0 = 256, H1 = 128, H2 = 64;
  const int BT = B * T;  // 32768 rows

  // workspace carve-out (~84 MB)
  char* ws = (char*)d_ws;
  size_t off = 0;
  auto carve = [&](size_t bytes) -> void* {
    void* p = ws + off;
    off = (off + bytes + 255) & ~(size_t)255;
    return p;
  };
  float*    xt  = (float*)carve((size_t)BT * H0 * sizeof(float));       // 32 MB
  float*    sp  = (float*)carve((size_t)BT * H0 * sizeof(float));       // 32 MB
  _Float16* yh  = (_Float16*)carve((size_t)BT * H0 * sizeof(_Float16)); // 16 MB
  _Float16* WT0 = (_Float16*)carve((size_t)D  * H0 * sizeof(_Float16));
  _Float16* WT1 = (_Float16*)carve((size_t)H0 * H1 * sizeof(_Float16));
  _Float16* WT2 = (_Float16*)carve((size_t)H1 * H2 * sizeof(_Float16));
  (void)ws_size; (void)in_sizes; (void)n_in; (void)out_size;

  // weight prep (transpose + half)
  k_wt<<<(D * H0 + 255) / 256, 256, 0, stream>>>(W0, WT0, D, H0);
  k_wt<<<(H0 * H1 + 255) / 256, 256, 0, stream>>>(W1, WT1, H0, H1);
  k_wt<<<(H1 * H2 + 255) / 256, 256, 0, stream>>>(W2, WT2, H1, H2);

  // layer 0: [BT,1024] x [1024,256]  (A read as f32 spikes, cvt in-regs)
  k_gemm<true><<<(BT / 32) * (H0 / 32) / 4, 128, 0, stream>>>(spikes, WT0, b0, xt, BT, H0, D);
  k_lif<<<B * H0 / 256, 256, 0, stream>>>(xt, sp, T, H0);
  k_ln<<<BT / 8, 256, 0, stream>>>(sp, ln1s, ln1b, yh, H0);

  // layer 1: [BT,256] x [256,128]
  k_gemm<false><<<(BT / 32) * (H1 / 32) / 4, 128, 0, stream>>>(yh, WT1, b1, xt, BT, H1, H0);
  k_lif<<<B * H1 / 256, 256, 0, stream>>>(xt, sp, T, H1);
  k_ln<<<BT / 8, 256, 0, stream>>>(sp, ln2s, ln2b, yh, H1);

  // layer 2: [BT,128] x [128,64]
  k_gemm<false><<<(BT / 32) * (H2 / 32) / 4, 128, 0, stream>>>(yh, WT2, b2, xt, BT, H2, H1);
  k_lif<<<B * H2 / 256, 256, 0, stream>>>(xt, sp, T, H2);

  // pooling + classifier head (out = [logits(2) | pooled(64) | mean(1)])
  k_pool<<<B, 256, 0, stream>>>(sp, out, T, H2);
  k_head<<<1, 64, 0, stream>>>(Wc, bc, out, B);
}
